// GeneralizedDenseMoE_16621523435695
// MI455X (gfx1250) — matrix operand order repaired and checked by reading, not compiled, not run on hardware
//
#include <hip/hip_runtime.h>
#include <hip/hip_bf16.h>
#include <math.h>

#define S_TOK 2048
#define DM    1024
#define NE    16
#define CAP   256   // 2*S/E

typedef __attribute__((ext_vector_type(16))) __bf16 v16bf;
typedef __attribute__((ext_vector_type(8)))  float  v8f;
typedef int v4i_ __attribute__((vector_size(16)));   // matches builtin param type

union Frag16 { uint4 q[2]; v16bf v; };
union Pack8  { unsigned short u[8]; uint4 q; };

#if __has_builtin(__builtin_amdgcn_global_load_async_to_lds_b128)
#define HAVE_ASYNC_LDS 1
#endif

__device__ __forceinline__ unsigned short f2bf(float f) {
  unsigned x = __float_as_uint(f);
  unsigned lsb = (x >> 16) & 1u;
  x += 0x7FFFu + lsb;              // round-to-nearest-even
  return (unsigned short)(x >> 16);
}

__device__ __forceinline__ void wait_async_lds() {
#if defined(HAVE_ASYNC_LDS)
#if __has_builtin(__builtin_amdgcn_s_wait_asynccnt)
  __builtin_amdgcn_s_wait_asynccnt(0);
#else
  asm volatile("s_wait_asynccnt 0x0" ::: "memory");
#endif
#endif
}

// ---------------- K1: gate logits = x @ wg^T (fp32, exact routing) ----------
__global__ void moe_gate_logits(const float* __restrict__ x,
                                const float* __restrict__ wg,
                                float* __restrict__ logits) {
  __shared__ float wgs[NE * DM];                  // 64 KB (WGP has 320 KB)
  int tid = threadIdx.x;
  for (int i = tid; i < NE * DM; i += 256) wgs[i] = wg[i];
  __syncthreads();
  int wave = tid >> 5, lane = tid & 31;
  int s = blockIdx.x * 8 + wave;                  // one wave per token
  const float4* xr = (const float4*)(x + (size_t)s * DM);
  float4 xv[8];
#pragma unroll
  for (int i = 0; i < 8; ++i) xv[i] = xr[lane + 32 * i];
  for (int e = 0; e < NE; ++e) {
    const float4* wr = (const float4*)(wgs + e * DM);
    float acc = 0.f;
#pragma unroll
    for (int i = 0; i < 8; ++i) {
      float4 w = wr[lane + 32 * i];
      acc += xv[i].x * w.x + xv[i].y * w.y + xv[i].z * w.z + xv[i].w * w.w;
    }
#pragma unroll
    for (int off = 16; off > 0; off >>= 1) acc += __shfl_xor(acc, off, 32);
    if (lane == 0) logits[s * NE + e] = acc;
  }
}

// ---------------- K2: top-2 gating + capacity ranks (single block) ----------
__global__ void moe_gating(const float* __restrict__ logits,
                           const float* __restrict__ gu,
                           int* __restrict__ idx1, int* __restrict__ idx2,
                           int* __restrict__ rank1, int* __restrict__ rank2,
                           int* __restrict__ count1,
                           float* __restrict__ g1raw, float* __restrict__ g2raw,
                           float* __restrict__ w1, float* __restrict__ w2,
                           int* __restrict__ slot1, int* __restrict__ slot2,
                           int* __restrict__ slot_token) {
  int tid = threadIdx.x;
  for (int t = tid; t < NE * CAP; t += 256) slot_token[t] = -1;
  for (int s = tid; s < S_TOK; s += 256) {
    float l[NE], ex[NE];
    float mx = -__builtin_inff();
#pragma unroll
    for (int e = 0; e < NE; ++e) { l[e] = logits[s * NE + e]; mx = fmaxf(mx, l[e]); }
    float sum = 0.f;
#pragma unroll
    for (int e = 0; e < NE; ++e) { ex[e] = expf(l[e] - mx); sum += ex[e]; }
    int i1 = 0; float best = l[0];
#pragma unroll
    for (int e = 1; e < NE; ++e) if (l[e] > best) { best = l[e]; i1 = e; }
    int i2 = -1; float best2 = -__builtin_inff();
#pragma unroll
    for (int e = 0; e < NE; ++e) {
      if (e == i1) continue;
      float gmb = -logf(-logf(gu[s * NE + e]));   // Gumbel(0,1)
      float z = l[e] + gmb;
      if (z > best2) { best2 = z; i2 = e; }
    }
    idx1[s] = i1; idx2[s] = i2;
    g1raw[s] = ex[i1] / sum; g2raw[s] = ex[i2] / sum;
  }
  __syncthreads();
  if (tid < NE) {                                  // sequential rank for mask1
    int e = tid, c = 0;
    for (int s = 0; s < S_TOK; ++s) if (idx1[s] == e) rank1[s] = c++;
    count1[e] = c;
  } else if (tid < 2 * NE) {                       // sequential rank for mask2
    int e = tid - NE, c = 0;
    for (int s = 0; s < S_TOK; ++s) if (idx2[s] == e) rank2[s] = c++;
  }
  __syncthreads();
  for (int s = tid; s < S_TOK; s += 256) {
    int i1 = idx1[s], i2 = idx2[s];
    int l1 = rank1[s];
    int l2 = rank2[s] + count1[i2];                // loc2 offset by full column sum
    bool k1 = l1 < CAP, k2 = l2 < CAP;
    float g1 = k1 ? g1raw[s] : 0.f;
    float g2 = k2 ? g2raw[s] : 0.f;
    float denom = fmaxf(g1 + g2, 1.1920929e-07f);  // finfo(f32).eps
    w1[s] = g1 / denom; w2[s] = g2 / denom;
    if (k1) { slot1[s] = i1 * CAP + l1; slot_token[i1 * CAP + l1] = s; } else slot1[s] = -1;
    if (k2) { slot2[s] = i2 * CAP + l2; slot_token[i2 * CAP + l2] = s; } else slot2[s] = -1;
  }
}

// ---------------- K3a: gather dispatched tokens -> bf16 A[E*C][DM] ----------
__global__ void moe_dispatch(const float* __restrict__ x,
                             const int* __restrict__ slot_token,
                             unsigned short* __restrict__ A) {
  int slot = blockIdx.x, tid = threadIdx.x;
  int tok = slot_token[slot];
  int d = tid * 8;
  Pack8 p;
  if (tok >= 0) {
    const float* xr = x + (size_t)tok * DM + d;
    float4 f0 = *(const float4*)(xr);
    float4 f1 = *(const float4*)(xr + 4);
    p.u[0] = f2bf(f0.x); p.u[1] = f2bf(f0.y); p.u[2] = f2bf(f0.z); p.u[3] = f2bf(f0.w);
    p.u[4] = f2bf(f1.x); p.u[5] = f2bf(f1.y); p.u[6] = f2bf(f1.z); p.u[7] = f2bf(f1.w);
  } else {
    p.q = uint4{0, 0, 0, 0};
  }
  *(uint4*)(A + (size_t)slot * DM + d) = p.q;
}

// ---------------- K3b: transpose-convert weights -> bf16 Wt[e][n][k] --------
__global__ void moe_wtrans(const float* __restrict__ w_exp,
                           unsigned short* __restrict__ Wt) {
  __shared__ float tile[32][33];
  int e = blockIdx.z;
  int nb = blockIdx.x * 32, kb = blockIdx.y * 32;
  int tx = threadIdx.x, ty = threadIdx.y;          // (32, 8)
  const float* src = w_exp + (size_t)e * DM * DM;
  for (int r = ty; r < 32; r += 8)
    tile[r][tx] = src[(size_t)(kb + r) * DM + nb + tx];   // coalesced over n
  __syncthreads();
  unsigned short* dst = Wt + (size_t)e * DM * DM;
  for (int r = ty; r < 32; r += 8)
    dst[(size_t)(nb + r) * DM + kb + tx] = f2bf(tile[tx][r]); // coalesced over k
}

// ---------------- K4: expert GEMM, WMMA + async-LDS double buffering --------
// Block = 8 waves = 256(M) x 64(N) stripe of one expert.  All waves share the
// same 64x32 bf16 B tile, staged in LDS *pre-swizzled into WMMA fragment
// order* via GLOBAL_LOAD_ASYNC_TO_LDS_B128 (double buffered, s_wait_asynccnt).
// Each wave: 32(M) x 64(N) -> 8 accumulators, 8 WMMAs per 32-wide k-step.

// Stage B tile for k-slice into lbuf (fragment order).  One b128 per thread.
__device__ __forceinline__ void stage_b_tile(const unsigned short* __restrict__ gB,
                                             unsigned short* __restrict__ lbuf,
                                             int t, int k) {
  int j = t >> 6;                 // which 16-wide N sub-tile (0..3)
  int lane = (t >> 1) & 31;       // destination fragment lane
  int half = t & 1;               // low/high 8 elements of the lane's 16
  int n = j * 16 + (lane & 15);
  int koff = k + ((lane >> 4) << 3) + half * 16;
  const unsigned short* src = gB + (size_t)n * DM + koff;
  unsigned short* dst = lbuf + ((j * 32 + lane) << 4) + half * 8;
#if defined(HAVE_ASYNC_LDS) && defined(__HIP_DEVICE_COMPILE__)
  __builtin_amdgcn_global_load_async_to_lds_b128(
      (__attribute__((address_space(1))) v4i_*)src,
      (__attribute__((address_space(3))) v4i_*)dst, 0, 0);
#else
  *(uint4*)dst = *(const uint4*)src;
#endif
}

__global__ void moe_expert_gemm(const unsigned short* __restrict__ A,
                                const unsigned short* __restrict__ Wt,
                                const float* __restrict__ b_exp,
                                float* __restrict__ eo) {
  __shared__ unsigned short btile[2][64 * 32];     // 2 x 4 KB, fragment order
  int tid = threadIdx.x;
  int wave = tid >> 5, lane = tid & 31;
  int e  = blockIdx.x >> 4;                        // 16 experts
  int nb = (blockIdx.x & 15) << 6;                 // 16 N-stripes of 64
  int mbase = wave << 5;                           // 32 rows per wave
  int lm = lane & 15, kh = lane >> 4;

  const unsigned short* gB = Wt + ((size_t)e * DM + nb) * DM;
  // A fragment rows (ISA 7.12.2 16-bit A layout: two 16B chunks / lane)
  const unsigned short* aRow0 = A + (size_t)(e * CAP + mbase + lm) * DM + kh * 8;
  const unsigned short* aRow1 = aRow0 + (size_t)16 * DM;

  v8f z = {0.f, 0.f, 0.f, 0.f, 0.f, 0.f, 0.f, 0.f};
  v8f acc[8];
#pragma unroll
  for (int j = 0; j < 8; ++j) acc[j] = z;

  // prologue: stage k=0 tile into buffer 0
  stage_b_tile(gB, &btile[0][0], tid, 0);
  wait_async_lds();
  __syncthreads();

#pragma unroll 2
  for (int i = 0; i < DM / 32; ++i) {
    int k = i * 32;
    int cur = i & 1, nxt = cur ^ 1;
    if (i + 1 < DM / 32)                           // prefetch next B tile
      stage_b_tile(gB, &btile[nxt][0], tid, k + 32);

    Frag16 a0, a1;
    a0.q[0] = *(const uint4*)(aRow0 + k);
    a0.q[1] = *(const uint4*)(aRow0 + k + 16);
    a1.q[0] = *(const uint4*)(aRow1 + k);
    a1.q[1] = *(const uint4*)(aRow1 + k + 16);

    // Preload all four B fragments so the 8 WMMAs can issue back-to-back
    // behind a single DScnt wait instead of fencing every pair.
    const unsigned short* lb = &btile[cur][0];
    Frag16 b[4];
#pragma unroll
    for (int j = 0; j < 4; ++j) {
      const unsigned short* fb = lb + ((j * 32 + lane) << 4);
      b[j].q[0] = *(const uint4*)(fb);
      b[j].q[1] = *(const uint4*)(fb + 8);
    }
#pragma unroll
    for (int j = 0; j < 4; ++j) {
      acc[j] = __builtin_amdgcn_wmma_f32_16x16x32_bf16(
          false, a0.v, false, b[j].v, (short)0, acc[j], false, false);
      acc[4 + j] = __builtin_amdgcn_wmma_f32_16x16x32_bf16(
          false, a1.v, false, b[j].v, (short)0, acc[4 + j], false, false);
    }
    wait_async_lds();
    __syncthreads();
  }

#pragma unroll
  for (int j = 0; j < 4; ++j) {
    int n = nb + j * 16 + lm;
    float bias = b_exp[e * DM + n];
#pragma unroll
    for (int r = 0; r < 8; ++r) {                  // C/D layout: VGPR r -> M
      int m0 = mbase + kh * 8 + r;
      eo[(size_t)(e * CAP + m0) * DM + n] = acc[j][r] + bias;
      eo[(size_t)(e * CAP + m0 + 16) * DM + n] = acc[4 + j][r] + bias;
    }
  }
}

// ---------------- K5: combine back to tokens --------------------------------
__global__ void moe_combine(const float* __restrict__ eo,
                            const float* __restrict__ w1, const float* __restrict__ w2,
                            const int* __restrict__ slot1, const int* __restrict__ slot2,
                            float* __restrict__ out) {
  int s = blockIdx.x, tid = threadIdx.x;           // 256 threads x float4 = 1024
  const float4* eo4 = (const float4*)eo;
  float4 v = {0.f, 0.f, 0.f, 0.f};
  int s1 = slot1[s], s2 = slot2[s];
  float a = w1[s], b = w2[s];
  if (s1 >= 0) {
    float4 t = eo4[(size_t)s1 * (DM / 4) + tid];
    v.x += a * t.x; v.y += a * t.y; v.z += a * t.z; v.w += a * t.w;
  }
  if (s2 >= 0) {
    float4 t = eo4[(size_t)s2 * (DM / 4) + tid];
    v.x += b * t.x; v.y += b * t.y; v.z += b * t.z; v.w += b * t.w;
  }
  ((float4*)out)[(size_t)s * (DM / 4) + tid] = v;
}

// ---------------- host ------------------------------------------------------
extern "C" void kernel_launch(void* const* d_in, const int* in_sizes, int n_in,
                              void* d_out, int out_size, void* d_ws, size_t ws_size,
                              hipStream_t stream) {
  const float* x     = (const float*)d_in[0];   // [1,2048,1024]
  const float* wg    = (const float*)d_in[1];   // [16,1024]
  const float* w_exp = (const float*)d_in[2];   // [16,1024,1024]
  const float* b_exp = (const float*)d_in[3];   // [16,1024]
  const float* gu    = (const float*)d_in[4];   // [2048,16]
  float* out = (float*)d_out;
  (void)in_sizes; (void)n_in; (void)out_size; (void)ws_size;

  char* ws = (char*)d_ws;
  size_t off = 0;
  auto carve = [&](size_t bytes) -> void* {
    void* p = ws + off;
    off += (bytes + 255) & ~(size_t)255;
    return p;
  };
  float* logits = (float*)carve(S_TOK * NE * 4);
  int* idx1   = (int*)carve(S_TOK * 4);
  int* idx2   = (int*)carve(S_TOK * 4);
  int* rank1  = (int*)carve(S_TOK * 4);
  int* rank2  = (int*)carve(S_TOK * 4);
  int* count1 = (int*)carve(NE * 4);
  float* g1raw = (float*)carve(S_TOK * 4);
  float* g2raw = (float*)carve(S_TOK * 4);
  float* w1s   = (float*)carve(S_TOK * 4);
  float* w2s   = (float*)carve(S_TOK * 4);
  int* slot1 = (int*)carve(S_TOK * 4);
  int* slot2 = (int*)carve(S_TOK * 4);
  int* stok  = (int*)carve(NE * CAP * 4);
  unsigned short* Abf = (unsigned short*)carve((size_t)NE * CAP * DM * 2);  // 8 MB
  unsigned short* Wt  = (unsigned short*)carve((size_t)NE * DM * DM * 2);   // 32 MB
  float* eo = (float*)carve((size_t)NE * CAP * DM * 4);                     // 16 MB

  moe_gate_logits<<<S_TOK / 8, 256, 0, stream>>>(x, wg, logits);
  moe_gating<<<1, 256, 0, stream>>>(logits, gu, idx1, idx2, rank1, rank2, count1,
                                    g1raw, g2raw, w1s, w2s, slot1, slot2, stok);
  moe_dispatch<<<NE * CAP, 128, 0, stream>>>(x, stok, Abf);
  moe_wtrans<<<dim3(DM / 32, DM / 32, NE), dim3(32, 8), 0, stream>>>(w_exp, Wt);
  moe_expert_gemm<<<NE * (DM / 64), 256, 0, stream>>>(Abf, Wt, b_exp, eo);
  moe_combine<<<S_TOK, 256, 0, stream>>>(eo, w1s, w2s, slot1, slot2, out);
}
